// VectorQuantizer_7799660609916
// MI455X (gfx1250) — matrix operand order repaired and checked by reading, not compiled, not run on hardware
//
#include <hip/hip_runtime.h>

// ---------------------------------------------------------------------------
// VQ-VAE vector quantizer for MI455X (gfx1250, wave32).
//   z_e        : [131072 x 64] fp32
//   embeddings : [1024   x 64] fp32
// Outputs (flat in d_out, fp32): z_q [N*64], vq_loss [1], indices-as-float [N]
//
// Strategy: argmin ||z-e||^2 == argmax (z.e - 0.5||e||^2).
// Scores via bf16 WMMA 16x16x32 with hi/lo error-compensated split
// (hi*hi + hi*lo + lo*hi, fp32 accumulate) => near-fp32 argmin accuracy at
// bf16 matrix throughput. Memory floor ~65MB @ 23.3TB/s; codebook (256KB bf16)
// stays L2/WGP$-resident.
// ---------------------------------------------------------------------------

#define VQ_N 131072
#define VQ_D 64
#define VQ_K 1024

typedef __attribute__((ext_vector_type(16))) __bf16 v16bf;
typedef __attribute__((ext_vector_type(8)))  float  v8f;

union BFrag { v16bf v; float4 f[2]; };

// d_ws layout (bytes):
#define WS_HESQ 0                         // 1024 floats: 0.5*||e_k||^2
#define WS_PART 4096                      // 1024 floats: per-block loss partials
#define WS_EHI  8192                      // 1024*64 bf16 (hi part)
#define WS_ELO  (8192 + VQ_K * VQ_D * 2)  // 1024*64 bf16 (lo part)

// ---------------------------------------------------------------------------
// Prep: half ||e||^2 and bf16 hi/lo split of the codebook. 1024 threads.
// ---------------------------------------------------------------------------
__global__ void vq_prep_kernel(const float* __restrict__ emb,
                               float* __restrict__ hesq,
                               __bf16* __restrict__ ehi,
                               __bf16* __restrict__ elo) {
  int k = blockIdx.x * blockDim.x + threadIdx.x;
  if (k >= VQ_K) return;
  const float* row = emb + (size_t)k * VQ_D;
  float s = 0.0f;
#pragma unroll 8
  for (int d = 0; d < VQ_D; ++d) {
    float x = row[d];
    s += x * x;
    __bf16 h = (__bf16)x;
    ehi[(size_t)k * VQ_D + d] = h;
    elo[(size_t)k * VQ_D + d] = (__bf16)(x - (float)h);
  }
  hesq[k] = 0.5f * s;
}

__device__ inline void split8_to(const float4 a, const float4 b,
                                 v16bf& hi, v16bf& lo, int base) {
  float v[8] = {a.x, a.y, a.z, a.w, b.x, b.y, b.z, b.w};
#pragma unroll
  for (int i = 0; i < 8; ++i) {
    __bf16 h = (__bf16)v[i];
    hi[base + i] = h;
    lo[base + i] = (__bf16)(v[i] - (float)h);
  }
}

// ---------------------------------------------------------------------------
// Main: 256 threads = 8 waves/block; each wave owns 16 rows, sweeps K=1024
// codes in 64 column tiles of 16 via 6 WMMAs/tile (2 K-chunks x 3 split terms).
// ---------------------------------------------------------------------------
__global__ void __launch_bounds__(256, 2)
vq_main_kernel(const float* __restrict__ z_e,
               const float* __restrict__ emb,
               const float* __restrict__ hesq,
               const __bf16* __restrict__ ehi,
               const __bf16* __restrict__ elo,
               float* __restrict__ out_zq,
               float* __restrict__ out_idx,
               float* __restrict__ partials) {
  const int lane = threadIdx.x & 31;
  const int w    = threadIdx.x >> 5;
  const int half = lane >> 4;    // 0: lanes 0-15, 1: lanes 16-31
  const int lr   = lane & 15;
  const int row0 = blockIdx.x * 128 + w * 16;

  // ---- Build A fragments (z_e rows, bf16 hi/lo) -------------------------
  // A 16x32 bf16 layout: lane row = lr (both halves), runs:
  //   elems 0..7  -> k = 32c + 8*half + i
  //   elems 8..15 -> k = 32c + 16 + 8*half + (i-8)
  const float* zrow = z_e + (size_t)(row0 + lr) * VQ_D;
  v16bf ahi[2], alo[2];
#pragma unroll
  for (int c = 0; c < 2; ++c) {
    const float4* zp = (const float4*)(zrow + 32 * c + 8 * half);
    float4 ra0 = zp[0], ra1 = zp[1];
    const float4* zq = (const float4*)(zrow + 32 * c + 16 + 8 * half);
    float4 rb0 = zq[0], rb1 = zq[1];
    split8_to(ra0, ra1, ahi[c], alo[c], 0);
    split8_to(rb0, rb1, ahi[c], alo[c], 8);
  }

  // ---- Sweep all 64 column tiles ----------------------------------------
  float bs[8];
  int   bidx[8];
#pragma unroll
  for (int i = 0; i < 8; ++i) { bs[i] = -3.0e38f; bidx[i] = 0; }

  for (int t = 0; t < 64; ++t) {
    const int col0 = t * 16;
    const float ce = hesq[col0 + lr];  // 0.5*||e||^2 for this lane's column
    const float4* erh = (const float4*)(ehi + (size_t)(col0 + lr) * VQ_D);
    const float4* erl = (const float4*)(elo + (size_t)(col0 + lr) * VQ_D);

    v8f cacc = {};
#pragma unroll
    for (int c = 0; c < 2; ++c) {
      // B 32x16 bf16 layout: lane col = lr, k runs = [32c + 8*half, +8) and [32c + 16 + 8*half, +8)
      BFrag bh, bl;
      const int o = (c * 32 + 8 * half) >> 3;  // float4 index (8 bf16 each)
      bh.f[0] = erh[o]; bh.f[1] = erh[o + 2];
      bl.f[0] = erl[o]; bl.f[1] = erl[o + 2];
      cacc = __builtin_amdgcn_wmma_f32_16x16x32_bf16(
          false, ahi[c], false, bh.v, (short)0, cacc, false, false);
      cacc = __builtin_amdgcn_wmma_f32_16x16x32_bf16(
          false, ahi[c], false, bl.v, (short)0, cacc, false, false);
      cacc = __builtin_amdgcn_wmma_f32_16x16x32_bf16(
          false, alo[c], false, bh.v, (short)0, cacc, false, false);
    }

    // C layout: VGPR i -> row (i + 8*half), col lr. Running argmax.
#pragma unroll
    for (int i = 0; i < 8; ++i) {
      float s = cacc[i] - ce;
      if (s > bs[i]) { bs[i] = s; bidx[i] = col0 + lr; }
    }
  }

  // ---- Cross-lane reduction (16 lanes share each row; stay in halves) ----
#pragma unroll
  for (int m = 1; m < 16; m <<= 1) {
#pragma unroll
    for (int i = 0; i < 8; ++i) {
      float ps = __shfl_xor(bs[i], m, 32);
      int   pi = __shfl_xor(bidx[i], m, 32);
      if (ps > bs[i] || (ps == bs[i] && pi < bidx[i])) { bs[i] = ps; bidx[i] = pi; }
    }
  }

  // ---- Write indices (as float): lane lr==i of each half owns row i+8*half
#pragma unroll
  for (int i = 0; i < 8; ++i) {
    if (lr == i) out_idx[row0 + i + 8 * half] = (float)bidx[i];
  }

  // ---- Gather z_q, write output, accumulate loss ------------------------
  float loss = 0.0f;
#pragma unroll
  for (int r = 0; r < 16; ++r) {
    const int src = (r >> 3) << 4;                  // lane 0 or 16
    const int bi  = __shfl(bidx[r & 7], src, 32);   // replicated within half
    const float2 e2 = *(const float2*)(emb + (size_t)bi * VQ_D + 2 * lane);
    const float2 z2 = *(const float2*)(z_e + (size_t)(row0 + r) * VQ_D + 2 * lane);
    *(float2*)(out_zq + (size_t)(row0 + r) * VQ_D + 2 * lane) = e2;
    float dx = e2.x - z2.x, dy = e2.y - z2.y;
    loss += dx * dx + dy * dy;
  }

  // Deterministic loss reduction: wave shuffle -> LDS (fixed order) -> partial
#pragma unroll
  for (int m = 1; m < 32; m <<= 1) loss += __shfl_xor(loss, m, 32);
  __shared__ float wl[8];
  if (lane == 0) wl[w] = loss;
  __syncthreads();
  if (threadIdx.x == 0) {
    float s = 0.0f;
#pragma unroll
    for (int i = 0; i < 8; ++i) s += wl[i];
    partials[blockIdx.x] = s;
  }
}

// ---------------------------------------------------------------------------
// Finalize: deterministic fixed-order sum of 1024 block partials.
// vq_loss = (1 + 0.25) * mean((z_q - z_e)^2)
// ---------------------------------------------------------------------------
__global__ void vq_finalize_kernel(const float* __restrict__ partials,
                                   float* __restrict__ out_loss) {
  if (threadIdx.x == 0 && blockIdx.x == 0) {
    float s = 0.0f;
    for (int i = 0; i < 1024; ++i) s += partials[i];
    out_loss[0] = 1.25f * s / (float)((size_t)VQ_N * VQ_D);
  }
}

extern "C" void kernel_launch(void* const* d_in, const int* in_sizes, int n_in,
                              void* d_out, int out_size, void* d_ws, size_t ws_size,
                              hipStream_t stream) {
  const float* z_e = (const float*)d_in[0];   // [131072 x 64]
  const float* emb = (const float*)d_in[1];   // [1024 x 64]

  char* ws = (char*)d_ws;
  float*  hesq     = (float*)(ws + WS_HESQ);
  float*  partials = (float*)(ws + WS_PART);
  __bf16* ehi      = (__bf16*)(ws + WS_EHI);
  __bf16* elo      = (__bf16*)(ws + WS_ELO);

  float* out      = (float*)d_out;
  float* out_zq   = out;                            // [N*64]
  float* out_loss = out + (size_t)VQ_N * VQ_D;      // [1]
  float* out_idx  = out_loss + 1;                   // [N]

  vq_prep_kernel<<<4, 256, 0, stream>>>(emb, hesq, ehi, elo);
  vq_main_kernel<<<VQ_N / 128, 256, 0, stream>>>(z_e, emb, hesq, ehi, elo,
                                                 out_zq, out_idx, partials);
  vq_finalize_kernel<<<1, 32, 0, stream>>>(partials, out_loss);
}
